// GlobalOut_69913477644864
// MI455X (gfx1250) — matrix operand order repaired and checked by reading, not compiled, not run on hardware
//
#include <hip/hip_runtime.h>
#include <math.h>

typedef __attribute__((ext_vector_type(16))) __bf16 v16bf;
typedef __attribute__((ext_vector_type(8)))  float  v8f;

union Frag {
    uint4 u[2];
    v16bf v;
};

__device__ __forceinline__ unsigned short f2bf(float f) {
    unsigned int u = __float_as_uint(f);
    unsigned int r = (u + 0x7FFFu + ((u >> 16) & 1u)) >> 16;   // RNE
    return (unsigned short)r;
}

// ---------------------------------------------------------------------------
// Kernel 1: fill output with quiet NaN (softmax of all -inf row == NaN)
// ---------------------------------------------------------------------------
__global__ void fill_nan_kernel(float* out, int n) {
    int i = blockIdx.x * blockDim.x + threadIdx.x;
    if (i < n) out[i] = __uint_as_float(0x7FC00000u);
}

// ---------------------------------------------------------------------------
// Kernel 2: gather x at idx and convert to bf16.  xg[m][ci], ci contiguous.
// ---------------------------------------------------------------------------
__global__ void gather_bf16_kernel(const float* __restrict__ x,
                                   const int* __restrict__ idx,
                                   unsigned short* __restrict__ xg) {
    int m = blockIdx.x;                 // 0..16383
    int l = idx[m];
    const float* src = x + (long)l * 1024;
    unsigned short* dst = xg + (long)m * 1024;
    for (int ci = threadIdx.x; ci < 1024; ci += blockDim.x)
        dst[ci] = f2bf(src[ci]);
}

// ---------------------------------------------------------------------------
// Kernel 3: repack w1 (co,ci,k) f32 -> Wt[co][k][ci] bf16 (ci contiguous)
// ---------------------------------------------------------------------------
__global__ void wprep_kernel(const float* __restrict__ w1,
                             unsigned short* __restrict__ Wt) {
    int i = blockIdx.x * blockDim.x + threadIdx.x;   // over 256*5120
    if (i >= 256 * 5120) return;
    int co = i / 5120;
    int rem = i % 5120;
    int k = rem / 1024;
    int ci = rem % 1024;
    Wt[i] = f2bf(w1[(long)co * 5120 + ci * 5 + k]);
}

// ---------------------------------------------------------------------------
// Kernel 4: conv1 as implicit GEMM with v_wmma_f32_16x16x32_bf16.
//   Block: 256 threads = 8 waves, arranged 4 (M) x 2 (N).
//   Block tile: M=128 positions x N=256 channels.
//   Wave tile:  M=32 (2 subtiles) x N=128 (8 subtiles) -> B frag reused 2x.
//   K loop: 32 ci-chunks (WMMA K=32) x 5 conv taps = 160 K-steps.
//   Weight tile staged with GLOBAL_LOAD_ASYNC_TO_LDS_B128 (ASYNCcnt path);
//   activation tile uses VGPR path (needs stripe-halo zero fill).
// Output: h[m][co] f32 = gelu(conv + bias)
// ---------------------------------------------------------------------------
__global__ void __launch_bounds__(256)
conv1_wmma_kernel(const unsigned short* __restrict__ xg,
                  const unsigned short* __restrict__ Wt,
                  const float* __restrict__ b1,
                  float* __restrict__ h) {
    __shared__ unsigned short Xs[132 * 32];   // 132 position rows (incl. +-2 halo) x 32 ci
    __shared__ unsigned short Ws[256 * 32];   // 256 out-channels x 32 ci

    const int tid  = threadIdx.x;
    const int wv   = tid >> 5;        // wave 0..7
    const int mw   = wv & 3;          // M quadrant (32 positions each)
    const int nw   = wv >> 2;         // N half (128 channels each)
    const int lane = tid & 31;
    const int lm   = lane & 15;
    const int half = lane >> 4;

    const int m0 = blockIdx.x * 128;        // first output position of tile
    const int s0 = (m0 >> 12) << 12;        // stripe start (4096-aligned)
    const int s1 = s0 + 4096;               // stripe end

    v8f acc[2][8];
#pragma unroll
    for (int s = 0; s < 2; ++s)
#pragma unroll
        for (int n = 0; n < 8; ++n) acc[s][n] = (v8f){};

    for (int cc = 0; cc < 32; ++cc) {       // ci-chunk (K=32 per WMMA)
        const int ci0 = cc * 32;
        __syncthreads();                    // previous compute done before Xs overwrite
        // stage activation tile: rows r=0..131 -> position p = m0-2+r (VGPR path, zero halo)
        for (int i = tid; i < 132 * 4; i += 256) {
            int r = i >> 2, seg = i & 3;
            int p = m0 - 2 + r;
            uint4 val = {0u, 0u, 0u, 0u};
            if (p >= s0 && p < s1)
                val = *(const uint4*)(xg + (long)p * 1024 + ci0 + seg * 8);
            *(uint4*)&Xs[r * 32 + seg * 8] = val;
        }
        for (int k = 0; k < 5; ++k) {       // conv tap
            __syncthreads();                // prev compute done before Ws overwrite; Xs visible
            // stage weight tile for (k, ci-chunk) via async global->LDS (16 KB)
#pragma unroll
            for (int it = 0; it < 4; ++it) {
                int i = tid + it * 256;     // 1024 x 16B chunks
                int co = i >> 2, seg = i & 3;
                unsigned lds_addr = (unsigned)(size_t)(&Ws[co * 32 + seg * 8]);
                unsigned long long gaddr =
                    (unsigned long long)(size_t)(Wt + (long)co * 5120 + k * 1024 + ci0 + seg * 8);
                asm volatile("global_load_async_to_lds_b128 %0, %1, off"
                             :: "v"(lds_addr), "v"(gaddr)
                             : "memory");
            }
            asm volatile("s_wait_asynccnt 0x0" ::: "memory");
            __syncthreads();                // tiles ready

            // A fragments (16x32 bf16), two M-subtiles for this wave.
            // Layout: lanes 0-15 hold K 0-7 (v0-3) and 16-23 (v4-7);
            //         lanes 16-31 hold K 8-15 and 24-31.
            int r0 = mw * 32 + lm + k;      // Xs row (halo shift +2 and tap offset k-2 cancel)
            int ka = half * 8;
            Frag af0, af1;
            af0.u[0] = *(const uint4*)&Xs[r0 * 32 + ka];
            af0.u[1] = *(const uint4*)&Xs[r0 * 32 + ka + 16];
            af1.u[0] = *(const uint4*)&Xs[(r0 + 16) * 32 + ka];
            af1.u[1] = *(const uint4*)&Xs[(r0 + 16) * 32 + ka + 16];

            // B fragments (32x16 bf16): lane n = N; lanes 0-15 K 0-7,16-23; 16-31 K 8-15,24-31.
            int kb = half * 8;
#pragma unroll
            for (int n = 0; n < 8; ++n) {
                int co = nw * 128 + n * 16 + lm;
                Frag bf;
                bf.u[0] = *(const uint4*)&Ws[co * 32 + kb];
                bf.u[1] = *(const uint4*)&Ws[co * 32 + kb + 16];
                acc[0][n] = __builtin_amdgcn_wmma_f32_16x16x32_bf16(
                    false, af0.v, false, bf.v, (short)0, acc[0][n], false, false);
                acc[1][n] = __builtin_amdgcn_wmma_f32_16x16x32_bf16(
                    false, af1.v, false, bf.v, (short)0, acc[1][n], false, false);
            }
        }
    }

    // Epilogue: bias + exact GELU, store f32 h[m][co].
    // C/D layout: lanes 0-15: VGPR r -> M=r, N=lane; lanes 16-31: M=8+r, N=lane-16.
#pragma unroll
    for (int n = 0; n < 8; ++n) {
        int col = nw * 128 + n * 16 + lm;
        float bias = b1[col];
#pragma unroll
        for (int s = 0; s < 2; ++s) {
#pragma unroll
            for (int rr = 0; rr < 8; ++rr) {
                int mrow = m0 + mw * 32 + s * 16 + (half ? rr + 8 : rr);
                float v = acc[s][n][rr] + bias;
                v = 0.5f * v * (1.0f + erff(v * 0.70710678118654752f));   // exact gelu
                h[(long)mrow * 256 + col] = v;
            }
        }
    }
}

// ---------------------------------------------------------------------------
// Kernel 5: conv2 (256 -> 3, K=5, global pad over 16384) + softmax + scatter.
// One thread per gathered position m.
// ---------------------------------------------------------------------------
__global__ void __launch_bounds__(256)
conv2_softmax_kernel(const float* __restrict__ h,
                     const float* __restrict__ w2,
                     const float* __restrict__ b2,
                     const int* __restrict__ idx,
                     float* __restrict__ out) {
    __shared__ float w2s[3 * 256 * 5];
    for (int i = threadIdx.x; i < 3 * 256 * 5; i += blockDim.x) w2s[i] = w2[i];
    __syncthreads();

    int m = blockIdx.x * blockDim.x + threadIdx.x;   // 0..16383
    float a0 = b2[0], a1 = b2[1], a2 = b2[2];
    for (int ci = 0; ci < 256; ++ci) {
        const float* wr0 = &w2s[0 * 1280 + ci * 5];
        const float* wr1 = &w2s[1 * 1280 + ci * 5];
        const float* wr2 = &w2s[2 * 1280 + ci * 5];
#pragma unroll
        for (int k = 0; k < 5; ++k) {
            int p = m + k - 2;
            if ((unsigned)p < 16384u) {
                float hv = h[(long)p * 256 + ci];
                a0 = fmaf(hv, wr0[k], a0);
                a1 = fmaf(hv, wr1[k], a1);
                a2 = fmaf(hv, wr2[k], a2);
            }
        }
    }
    float mx = fmaxf(a0, fmaxf(a1, a2));
    float e0 = __expf(a0 - mx), e1 = __expf(a1 - mx), e2 = __expf(a2 - mx);
    float inv = 1.0f / (e0 + e1 + e2);
    long l = idx[m];
    out[l * 3 + 0] = e0 * inv;
    out[l * 3 + 1] = e1 * inv;
    out[l * 3 + 2] = e2 * inv;
}

// ---------------------------------------------------------------------------
extern "C" void kernel_launch(void* const* d_in, const int* in_sizes, int n_in,
                              void* d_out, int out_size, void* d_ws, size_t ws_size,
                              hipStream_t stream) {
    const float* x   = (const float*)d_in[0];   // (1,131072,1024)
    const float* w1  = (const float*)d_in[1];   // (256,1024,5)
    const float* b1  = (const float*)d_in[2];   // (256)
    const float* w2  = (const float*)d_in[3];   // (3,256,5)
    const float* b2  = (const float*)d_in[4];   // (3)
    const int*   idx = (const int*)d_in[5];     // (16384)
    float* out = (float*)d_out;                 // (1,131072,3) = 393216 f32

    // workspace layout
    const size_t XG_BYTES = (size_t)16384 * 1024 * 2;   // 33.55 MB bf16
    const size_t WT_BYTES = (size_t)256 * 5120 * 2;     //  2.62 MB bf16
    char* ws = (char*)d_ws;
    unsigned short* xg = (unsigned short*)ws;
    unsigned short* Wt = (unsigned short*)(ws + XG_BYTES);
    float*          hb = (float*)(ws + XG_BYTES + WT_BYTES);   // 16.78 MB f32

    // 1) fill output with NaN (matches softmax of all -inf rows)
    fill_nan_kernel<<<(393216 + 255) / 256, 256, 0, stream>>>(out, 393216);
    // 2) gather + convert to bf16
    gather_bf16_kernel<<<16384, 256, 0, stream>>>(x, idx, xg);
    // 3) repack weights to bf16 [co][k][ci]
    wprep_kernel<<<(256 * 5120 + 255) / 256, 256, 0, stream>>>(w1, Wt);
    // 4) conv1 as WMMA implicit GEMM (M=16384, N=256, K=5120) + bias + gelu
    conv1_wmma_kernel<<<16384 / 128, 256, 0, stream>>>(xg, Wt, b1, hb);
    // 5) conv2 + softmax + scatter
    conv2_softmax_kernel<<<16384 / 256, 256, 0, stream>>>(hb, w2, b2, idx, out);
}
